// Predictor_12575664242960
// MI455X (gfx1250) — compile-verified, matrix operand order checked by reading, not run on hardware
//
#include <hip/hip_runtime.h>
#include <cstdint>
#include <cstddef>

typedef __bf16 bf16;
typedef __attribute__((ext_vector_type(16))) bf16  v16bf;
typedef __attribute__((ext_vector_type(8)))  float v8f;

constexpr int kB = 4, kT = 1024, kV = 32000, kD = 384, kH = 6, kHS = 64, kL = 6, kDFF = 1536;
constexpr int kM = kB * kT;  // 4096 rows

// ---------------------------------------------------------------------------
// Fragment loader: CDNA5 16-bit A/B layout. Lane holds row (M or N) = lane%16;
// elements 0..7  = K [8*half .. 8*half+7], elements 8..15 = K [16+8*half ..].
// Both runs are 16B contiguous -> two ds_load_b128.
// ---------------------------------------------------------------------------
__device__ __forceinline__ v16bf ld_frag16(const bf16* row, int half) {
  union { uint4 u[2]; v16bf v; } f;
  f.u[0] = *reinterpret_cast<const uint4*>(row + 8 * half);
  f.u[1] = *reinterpret_cast<const uint4*>(row + 16 + 8 * half);
  return f.v;
}

// CDNA5 async global->LDS 16B copy (ASYNCcnt-tracked DMA, bypasses VGPRs).
// Generic shared pointer's low 32 bits are the LDS byte offset (ISA 10.2).
__device__ __forceinline__ void async_cp16(void* lds, const void* gsrc) {
  uint32_t l = (uint32_t)(uintptr_t)lds;
  asm volatile("global_load_async_to_lds_b128 %0, %1, off"
               :: "v"(l), "v"(gsrc) : "memory");
}
__device__ __forceinline__ void wait_async0() {
  asm volatile("s_wait_asynccnt 0x0" ::: "memory");
}

// ---------------------------------------------------------------------------
// Generic bf16 GEMM: C[M,N] = A[M,K] (row-major bf16) x Bt[N,K] (row-major,
// i.e. B pre-transposed).  Block tile 128x128, BK=32, 8 waves (256 threads)
// in a 4x2 grid; each wave computes 32x64 as 2x4 WMMA 16x16 tiles (8 WMMA /
// K-step).  LDS double-buffered and filled with async global->LDS DMA so the
// next tile's fill overlaps this tile's WMMAs.
// OUT_MODE: 0 = f32 row-major, 1 = bf16 row-major, 2 = bf16 transposed
// ([b][n][t] with M = b*kT + t) used to produce V^T for attention.
// ---------------------------------------------------------------------------
template<int OUT_MODE, bool BIAS, bool RELU, bool RESID>
__global__ __launch_bounds__(256)
void gemm_bf16_k(const bf16* __restrict__ A, const bf16* __restrict__ Bt,
                 const float* __restrict__ bias, const float* __restrict__ resid,
                 void* __restrict__ outp, int M, int N, int K)
{
  constexpr int LS = 40;  // padded LDS row stride (banks spread, 16B aligned)
  // rows 0..127 = A tile, rows 128..255 = B tile; two buffers
  __shared__ __align__(16) bf16 smbuf[2][256 * LS];
  const int tid   = threadIdx.x;
  const int lane  = tid & 31;
  const int w     = tid >> 5;
  const int waveM = w >> 1, waveN = w & 1;   // 4 x 2 wave grid
  const int laneH = lane & 15;
  const int half  = lane >> 4;
  const int mBlock = blockIdx.y * 128;
  const int nBlock = blockIdx.x * 128;

  // 1024 x 16B chunks per stage, 4 per thread
  auto issue_stage = [&](int buf, int kt) {
    #pragma unroll
    for (int i = 0; i < 4; ++i) {
      const int chunk = tid + i * 256;
      const int r  = chunk >> 2;          // 0..255
      const int cg = (chunk & 3) * 8;
      const bf16* src = (r < 128)
          ? A  + (size_t)(mBlock + r)         * K + kt * 32 + cg
          : Bt + (size_t)(nBlock + (r - 128)) * K + kt * 32 + cg;
      async_cp16(&smbuf[buf][r * LS + cg], src);
    }
  };

  v8f acc[2][4] = {};
  const int kIters = K >> 5;
  issue_stage(0, 0);
  int cur = 0;
  for (int kt = 0; kt < kIters; ++kt) {
    wait_async0();        // my stage-`cur` DMAs landed
    __syncthreads();      // everyone's landed; buffer cur^1 free to refill
    if (kt + 1 < kIters) issue_stage(cur ^ 1, kt + 1);
    const bf16* Abase = &smbuf[cur][0];
    const bf16* Bbase = &smbuf[cur][128 * LS];
    v16bf af[2], bfr[4];
    #pragma unroll
    for (int tm = 0; tm < 2; ++tm)
      af[tm]  = ld_frag16(Abase + (waveM * 32 + tm * 16 + laneH) * LS, half);
    #pragma unroll
    for (int tn = 0; tn < 4; ++tn)
      bfr[tn] = ld_frag16(Bbase + (waveN * 64 + tn * 16 + laneH) * LS, half);
    #pragma unroll
    for (int tm = 0; tm < 2; ++tm)
      #pragma unroll
      for (int tn = 0; tn < 4; ++tn)
        acc[tm][tn] = __builtin_amdgcn_wmma_f32_16x16x32_bf16(
            false, af[tm], false, bfr[tn], (short)0, acc[tm][tn], false, false);
    cur ^= 1;
  }

  #pragma unroll
  for (int tm = 0; tm < 2; ++tm) {
    #pragma unroll
    for (int tn = 0; tn < 4; ++tn) {
      const int n = nBlock + waveN * 64 + tn * 16 + laneH;
      float bv = 0.f;
      if (BIAS) bv = bias[n];
      const int mBase = mBlock + waveM * 32 + tm * 16 + 8 * half;  // C: M = i + 8*half
      if (OUT_MODE == 2) {
        union { bf16 h[8]; uint4 u; } pk;
        #pragma unroll
        for (int i = 0; i < 8; ++i) pk.h[i] = (bf16)acc[tm][tn][i];
        const int bb = mBase / kT, t0 = mBase % kT;  // 8 consecutive t -> one b128 store
        bf16* o = (bf16*)outp + (size_t)bb * N * kT + (size_t)n * kT + t0;
        *reinterpret_cast<uint4*>(o) = pk.u;
      } else {
        #pragma unroll
        for (int i = 0; i < 8; ++i) {
          float v = acc[tm][tn][i] + bv;
          if (RELU) v = fmaxf(v, 0.f);
          const size_t idx = (size_t)(mBase + i) * N + n;
          if (RESID) v += resid[idx];
          if (OUT_MODE == 0) ((float*)outp)[idx] = v;
          else               ((bf16*)outp)[idx]  = (bf16)v;
        }
      }
    }
  }
}

// ---------------------------------------------------------------------------
// Flash-style attention, one wave per (b, h, 16 query rows).
// NOTE: reference multiplies scores by tril (zeros, NOT -inf) pre-softmax, so
// every key contributes exp(masked 0) to the denominator and attn*v sums over
// ALL keys -> stream all key blocks with masked score = 0.
// q,k: [B,T,H,HS] bf16; vT: [B,H,HS,T] bf16; av: [B,T,H,HS] bf16.
// ---------------------------------------------------------------------------
__global__ __launch_bounds__(32)
void attn_k(const bf16* __restrict__ q, const bf16* __restrict__ kk,
            const bf16* __restrict__ vT, bf16* __restrict__ av)
{
  __shared__ __align__(16) bf16 Ps[16 * 40];  // P tile staging (C -> A layout)
  const int lane = threadIdx.x;
  const int laneH = lane & 15, half = lane >> 4;
  const int bh = blockIdx.y;
  const int b = bh / kH, h = bh % kH;
  const int qBase = blockIdx.x * 16;
  const float scale = 0.125f;  // 1/sqrt(64)

  const bf16* qrow = q + (((size_t)b * kT + qBase + laneH) * kH + h) * kHS;
  v16bf qa[2];
  #pragma unroll
  for (int ks = 0; ks < 2; ++ks) qa[ks] = ld_frag16(qrow + ks * 32, half);

  v8f O[4] = {};
  float mrow[8], lrow[8];
  #pragma unroll
  for (int i = 0; i < 8; ++i) { mrow[i] = -3.0e38f; lrow[i] = 0.f; }

  for (int kb = 0; kb < kT / 32; ++kb) {
    // S[16x32] = Q(16x64) x K^T(64x32): 4 WMMA
    v8f s[2] = {};
    #pragma unroll
    for (int tn = 0; tn < 2; ++tn) {
      const int key = kb * 32 + tn * 16 + laneH;
      const bf16* krow = kk + (((size_t)b * kT + key) * kH + h) * kHS;
      #pragma unroll
      for (int ks = 0; ks < 2; ++ks) {
        v16bf kf = ld_frag16(krow + ks * 32, half);
        s[tn] = __builtin_amdgcn_wmma_f32_16x16x32_bf16(false, qa[ks], false, kf, (short)0, s[tn], false, false);
      }
    }
    float sv[2][8];
    #pragma unroll
    for (int tn = 0; tn < 2; ++tn) {
      const int key = kb * 32 + tn * 16 + laneH;
      #pragma unroll
      for (int i = 0; i < 8; ++i) {
        const int t = qBase + i + 8 * half;
        sv[tn][i] = (key <= t) ? s[tn][i] * scale : 0.f;  // tril zeroing, then scale
      }
    }
    // online softmax; row M lives across the 16 lanes of a half-wave
    #pragma unroll
    for (int i = 0; i < 8; ++i) {
      float rmax = fmaxf(sv[0][i], sv[1][i]);
      #pragma unroll
      for (int off = 8; off >= 1; off >>= 1) rmax = fmaxf(rmax, __shfl_xor(rmax, off, 32));
      const float mnew = fmaxf(mrow[i], rmax);
      const float fsc  = __expf(mrow[i] - mnew);
      const float p0 = __expf(sv[0][i] - mnew);
      const float p1 = __expf(sv[1][i] - mnew);
      float rs = p0 + p1;
      #pragma unroll
      for (int off = 8; off >= 1; off >>= 1) rs += __shfl_xor(rs, off, 32);
      lrow[i] = lrow[i] * fsc + rs;
      mrow[i] = mnew;
      #pragma unroll
      for (int ot = 0; ot < 4; ++ot) O[ot][i] *= fsc;
      Ps[(i + 8 * half) * 40 + laneH]      = (bf16)p0;   // C-layout stash
      Ps[(i + 8 * half) * 40 + 16 + laneH] = (bf16)p1;
    }
    __syncthreads();
    v16bf pa = ld_frag16(&Ps[laneH * 40], half);         // re-read as A-frag
    // O += P(16x32) x V(32x64): 4 WMMA, V^T rows contiguous in t
    #pragma unroll
    for (int ot = 0; ot < 4; ++ot) {
      const bf16* vrow = vT + (((size_t)b * kH + h) * kHS + ot * 16 + laneH) * kT + kb * 32;
      v16bf vf = ld_frag16(vrow, half);
      O[ot] = __builtin_amdgcn_wmma_f32_16x16x32_bf16(false, pa, false, vf, (short)0, O[ot], false, false);
    }
    __syncthreads();
  }
  #pragma unroll
  for (int i = 0; i < 8; ++i) {
    const float inv = 1.0f / lrow[i];
    const int t = qBase + i + 8 * half;
    bf16* orow = av + (((size_t)b * kT + t) * kH + h) * kHS;
    #pragma unroll
    for (int ot = 0; ot < 4; ++ot) orow[ot * 16 + laneH] = (bf16)(O[ot][i] * inv);
  }
}

// ---------------------------------------------------------------------------
// LayerNorm over D=384, f32 in -> bf16 out, one block (128 thr) per row.
// ---------------------------------------------------------------------------
__global__ __launch_bounds__(128)
void layernorm_k(const float* __restrict__ x, const float* __restrict__ g,
                 const float* __restrict__ bta, bf16* __restrict__ out)
{
  __shared__ float r1[128], r2[128];
  const int row = blockIdx.x, tid = threadIdx.x;
  const float* xr = x + (size_t)row * kD;
  const float v0 = xr[tid], v1 = xr[tid + 128], v2 = xr[tid + 256];
  r1[tid] = v0 + v1 + v2;
  r2[tid] = v0 * v0 + v1 * v1 + v2 * v2;
  __syncthreads();
  for (int st = 64; st > 0; st >>= 1) {
    if (tid < st) { r1[tid] += r1[tid + st]; r2[tid] += r2[tid + st]; }
    __syncthreads();
  }
  const float mean = r1[0] * (1.0f / kD);
  const float var  = r2[0] * (1.0f / kD) - mean * mean;
  const float rstd = rsqrtf(var + 1e-5f);
  bf16* orow = out + (size_t)row * kD;
  orow[tid]       = (bf16)((v0 - mean) * rstd * g[tid]       + bta[tid]);
  orow[tid + 128] = (bf16)((v1 - mean) * rstd * g[tid + 128] + bta[tid + 128]);
  orow[tid + 256] = (bf16)((v2 - mean) * rstd * g[tid + 256] + bta[tid + 256]);
}

__global__ __launch_bounds__(128)
void embed_k(const int* __restrict__ tok, const float* __restrict__ tokTab,
             const float* __restrict__ posTab, float* __restrict__ x)
{
  const int row = blockIdx.x, tid = threadIdx.x;
  const int t = row & (kT - 1);
  const int id = tok[row];
  const float* tr = tokTab + (size_t)id * kD;
  const float* pr = posTab + (size_t)t * kD;
  float* xr = x + (size_t)row * kD;
  #pragma unroll
  for (int j = 0; j < 3; ++j) { const int d = tid + j * 128; xr[d] = tr[d] + pr[d]; }
}

// f32 -> bf16 with transpose: in slice [R,C] -> out[c*outStride + r].
__global__ void cvt_tr_k(const float* __restrict__ in, bf16* __restrict__ out,
                         int R, int C, long inZ, long outL, long outH, int perL, int outStride)
{
  const int z = blockIdx.z;
  const float* src = in + (size_t)z * inZ;
  bf16* dst = out + (size_t)(z / perL) * outL + (size_t)(z % perL) * outH;
  const int c = blockIdx.x * 16 + threadIdx.x;
  const int r = blockIdx.y * 16 + threadIdx.y;
  if (r < R && c < C)
    dst[(size_t)c * outStride + r] = (bf16)src[(size_t)r * C + c];
}

__global__ void zero_k(float* p) { if (threadIdx.x == 0 && blockIdx.x == 0) *p = 0.f; }

// Per-row online logsumexp over V=32000, then atomic mean of (lse - gold).
__global__ __launch_bounds__(256)
void loss_k(const float* __restrict__ logits, const int* __restrict__ tgt,
            float* __restrict__ lossp)
{
  __shared__ float sm[256], ssum[256];
  const int row = blockIdx.x, tid = threadIdx.x;
  const float* lr = logits + (size_t)row * kV;
  float m = -3.0e38f, s = 0.f;
  for (int j = tid; j < kV; j += 256) {
    const float xv = lr[j];
    if (xv > m) { s = s * __expf(m - xv) + 1.f; m = xv; }
    else        { s += __expf(xv - m); }
  }
  sm[tid] = m; ssum[tid] = s;
  __syncthreads();
  for (int st = 128; st > 0; st >>= 1) {
    if (tid < st) {
      const float m2 = sm[tid + st], s2 = ssum[tid + st];
      const float mm = fmaxf(sm[tid], m2);
      ssum[tid] = ssum[tid] * __expf(sm[tid] - mm) + s2 * __expf(m2 - mm);
      sm[tid] = mm;
    }
    __syncthreads();
  }
  if (tid == 0) {
    const float lse  = sm[0] + __logf(ssum[0]);
    const float gold = lr[tgt[row]];
    atomicAdd(lossp, (lse - gold) * (1.0f / (float)kM));
  }
}

// ---------------------------------------------------------------------------
extern "C" void kernel_launch(void* const* d_in, const int* in_sizes, int n_in,
                              void* d_out, int out_size, void* d_ws, size_t ws_size,
                              hipStream_t stream)
{
  (void)in_sizes; (void)n_in; (void)out_size; (void)ws_size;
  const int*   token_ids = (const int*)d_in[0];
  const int*   targets   = (const int*)d_in[1];
  const float* tok_table = (const float*)d_in[2];
  const float* pos_table = (const float*)d_in[3];
  const float* ln1_g = (const float*)d_in[4];
  const float* ln1_b = (const float*)d_in[5];
  const float* Wq = (const float*)d_in[6];
  const float* Wk = (const float*)d_in[7];
  const float* Wv = (const float*)d_in[8];
  const float* Wo = (const float*)d_in[9];
  const float* ln2_g = (const float*)d_in[10];
  const float* ln2_b = (const float*)d_in[11];
  const float* W1 = (const float*)d_in[12];
  const float* b1 = (const float*)d_in[13];
  const float* W2 = (const float*)d_in[14];
  const float* b2 = (const float*)d_in[15];
  const float* lnf_g = (const float*)d_in[16];
  const float* lnf_b = (const float*)d_in[17];
  const float* Wlm = (const float*)d_in[18];
  const float* blm = (const float*)d_in[19];

  float* logits = (float*)d_out;
  float* lossp  = logits + (size_t)kM * kV;

  // ---- workspace carve (~80 MB) ----
  char* wptr = (char*)d_ws;
  auto carve = [&](size_t bytes) -> void* {
    void* p = (void*)wptr;
    wptr += (bytes + 255) & ~(size_t)255;
    return p;
  };
  const size_t szQKV = (size_t)kL * kH * kHS * kD;   // per-matrix elems
  const size_t szFF  = (size_t)kL * kDFF * kD;
  bf16* wq_t  = (bf16*)carve(szQKV * 2);
  bf16* wk_t  = (bf16*)carve(szQKV * 2);
  bf16* wv_t  = (bf16*)carve(szQKV * 2);
  bf16* wo_t  = (bf16*)carve(szQKV * 2);
  bf16* w1_t  = (bf16*)carve(szFF * 2);
  bf16* w2_t  = (bf16*)carve(szFF * 2);
  bf16* wlm_t = (bf16*)carve((size_t)kV * kD * 2);
  float* x    = (float*)carve((size_t)kM * kD * 4);
  bf16* hbf   = (bf16*)carve((size_t)kM * kD * 2);
  bf16* qb    = (bf16*)carve((size_t)kM * kD * 2);
  bf16* kb    = (bf16*)carve((size_t)kM * kD * 2);
  bf16* vtb   = (bf16*)carve((size_t)kM * kD * 2);
  bf16* avb   = (bf16*)carve((size_t)kM * kD * 2);
  bf16* ffb   = (bf16*)carve((size_t)kM * kDFF * 2);

  const dim3 blk16(16, 16);
  // Wq/Wk/Wv: (L,H,D,HS) -> [l][h*HS+e][d]
  cvt_tr_k<<<dim3(kHS/16, kD/16, kL*kH), blk16, 0, stream>>>(Wq, wq_t, kD, kHS,
      (long)kD*kHS, (long)kH*kHS*kD, (long)kHS*kD, kH, kD);
  cvt_tr_k<<<dim3(kHS/16, kD/16, kL*kH), blk16, 0, stream>>>(Wk, wk_t, kD, kHS,
      (long)kD*kHS, (long)kH*kHS*kD, (long)kHS*kD, kH, kD);
  cvt_tr_k<<<dim3(kHS/16, kD/16, kL*kH), blk16, 0, stream>>>(Wv, wv_t, kD, kHS,
      (long)kD*kHS, (long)kH*kHS*kD, (long)kHS*kD, kH, kD);
  // Wo: (L,H,HS,D) -> [l][d][h*HS+e]
  cvt_tr_k<<<dim3(kD/16, kHS/16, kL*kH), blk16, 0, stream>>>(Wo, wo_t, kHS, kD,
      (long)kHS*kD, (long)kD*kH*kHS, (long)kHS, kH, kH*kHS);
  // W1: (L,D,DFF) -> [l][dff][d] ; W2: (L,DFF,D) -> [l][d][dff]
  cvt_tr_k<<<dim3(kDFF/16, kD/16, kL), blk16, 0, stream>>>(W1, w1_t, kD, kDFF,
      (long)kD*kDFF, (long)kDFF*kD, 0, 1, kD);
  cvt_tr_k<<<dim3(kD/16, kDFF/16, kL), blk16, 0, stream>>>(W2, w2_t, kDFF, kD,
      (long)kDFF*kD, (long)kD*kDFF, 0, 1, kDFF);
  // Wlm: (D,V) -> [v][d]
  cvt_tr_k<<<dim3(kV/16, kD/16, 1), blk16, 0, stream>>>(Wlm, wlm_t, kD, kV,
      0, 0, 0, 1, kD);

  embed_k<<<kM, 128, 0, stream>>>(token_ids, tok_table, pos_table, x);

  const dim3 gD(kD / 128, kM / 128);      // (3, 32)
  const dim3 gFF(kDFF / 128, kM / 128);   // (12, 32)
  for (int l = 0; l < kL; ++l) {
    layernorm_k<<<kM, 128, 0, stream>>>(x, ln1_g + l*kD, ln1_b + l*kD, hbf);
    const size_t oQ = (size_t)l * kH * kHS * kD;
    gemm_bf16_k<1,false,false,false><<<gD, 256, 0, stream>>>(hbf, wq_t + oQ, nullptr, nullptr, qb,  kM, kD, kD);
    gemm_bf16_k<1,false,false,false><<<gD, 256, 0, stream>>>(hbf, wk_t + oQ, nullptr, nullptr, kb,  kM, kD, kD);
    gemm_bf16_k<2,false,false,false><<<gD, 256, 0, stream>>>(hbf, wv_t + oQ, nullptr, nullptr, vtb, kM, kD, kD);
    attn_k<<<dim3(kT/16, kB*kH), 32, 0, stream>>>(qb, kb, vtb, avb);
    gemm_bf16_k<0,false,false,true><<<gD, 256, 0, stream>>>(avb, wo_t + oQ, nullptr, x, x, kM, kD, kD);
    layernorm_k<<<kM, 128, 0, stream>>>(x, ln2_g + l*kD, ln2_b + l*kD, hbf);
    const size_t oF = (size_t)l * kDFF * kD;
    gemm_bf16_k<1,true,true,false><<<gFF, 256, 0, stream>>>(hbf, w1_t + oF, b1 + l*kDFF, nullptr, ffb, kM, kDFF, kD);
    gemm_bf16_k<0,true,false,true><<<gD, 256, 0, stream>>>(ffb, w2_t + oF, b2 + l*kD, x, x, kM, kD, kDFF);
  }

  layernorm_k<<<kM, 128, 0, stream>>>(x, lnf_g, lnf_b, hbf);
  gemm_bf16_k<0,true,false,false><<<dim3(kV/128, kM/128), 256, 0, stream>>>(
      hbf, wlm_t, blm, nullptr, logits, kM, kV, kD);

  zero_k<<<1, 32, 0, stream>>>(lossp);
  loss_k<<<kM, 256, 0, stream>>>(logits, targets, lossp);
}
